// LSTM_49409303773380
// MI455X (gfx1250) — matrix operand order repaired and checked by reading, not compiled, not run on hardware
//
#include <hip/hip_runtime.h>
#include <hip/hip_bf16.h>

typedef __bf16 v16bf __attribute__((ext_vector_type(16)));
typedef float  v8f   __attribute__((ext_vector_type(8)));

#define SEQ 512
#define DIM 256
#define HD  1024
#define NC  10
#define BS  128

#define KC_X 8    // DIM/32  K-chunks from x
#define KC_H 32   // HD/32   K-chunks from h
#define MT   8    // BS/16   m-tiles
#define JT   64   // HD/16   j-tiles

#define NGROUPS 5           // group 0 = x panel (8 chunks), groups 1..4 = h panel
#define CHUNKS_PER_GROUP 8
#define CHUNK_LDS_BYTES 4096u     // 4 gates * 32 lanes * 16 bf16 * 2B
#define GROUP_LDS_BYTES 32768u    // 8 chunks

// element counts
#define AX_ELEMS (SEQ * KC_X * MT * 32 * 16)   // 16,777,216 bf16
#define BH_ELEMS (KC_H * 4 * JT * 32 * 16)     //  4,194,304 bf16
#define BX_ELEMS (KC_X * 4 * JT * 32 * 16)     //  1,048,576 bf16
#define AH_ELEMS (KC_H * MT * 32 * 16)         //    131,072 bf16 (one buffer)
#define CH_ELEMS (BS * HD)                     //    131,072 f32

// workspace byte offsets (all 4KB-aligned)
#define BH_OFF 33554432ull
#define BX_OFF 41943040ull
#define AH_OFF 44040192ull   // two ping-pong buffers of 262144 B
#define C_OFF  44564480ull
#define H_OFF  45088768ull

__device__ __forceinline__ float sigf(float x) {
    return 1.0f / (1.0f + __expf(-x));
}

// CDNA5 async global->LDS copy (ASYNCcnt-tracked, no VGPR data).
__device__ __forceinline__ void async_g2l_b128(unsigned lds_addr, const void* gaddr) {
    asm volatile("global_load_async_to_lds_b128 %0, %1, off"
                 :: "v"(lds_addr), "v"(gaddr) : "memory");
}
__device__ __forceinline__ void wait_asynccnt0() {
    asm volatile("s_wait_asynccnt 0x0" ::: "memory");
}

// ---- pack x (B,T,D) f32 -> A-fragment bf16 layout [t][kc][mt][lane][e] ----
__global__ void pack_x_kernel(const float* __restrict__ x, __bf16* __restrict__ ax) {
    unsigned id   = blockIdx.x * 256u + threadIdx.x;          // < AX_ELEMS
    unsigned e    = id & 15u;
    unsigned lane = (id >> 4) & 31u;
    unsigned mt   = (id >> 9) & 7u;
    unsigned kc   = (id >> 12) & 7u;
    unsigned t    = id >> 15;
    unsigned d    = kc * 32u + e + 8u * (e >> 3) + 8u * (lane >> 4);
    unsigned b    = mt * 16u + (lane & 15u);
    ax[id] = (__bf16)x[(b * SEQ + t) * DIM + d];
}

// ---- pack 4 weight mats (K,1024) f32 -> B-fragment bf16 [kc][gate][jt][lane][e] ----
__global__ void pack_wb_kernel(const float* __restrict__ Wg, const float* __restrict__ Wi,
                               const float* __restrict__ Wf, const float* __restrict__ Wo,
                               __bf16* __restrict__ out) {
    unsigned id   = blockIdx.x * 256u + threadIdx.x;
    unsigned e    = id & 15u;
    unsigned lane = (id >> 4) & 31u;
    unsigned jt   = (id >> 9) & 63u;
    unsigned g    = (id >> 15) & 3u;
    unsigned kc   = id >> 17;
    unsigned k    = kc * 32u + e + 8u * (e >> 3) + 8u * (lane >> 4);
    unsigned j    = jt * 16u + (lane & 15u);
    const float* W = (g == 0u) ? Wg : (g == 1u) ? Wi : (g == 2u) ? Wf : Wo;
    out[id] = (__bf16)W[k * HD + j];
}

// ---- zero h0 (packed) and c ----
__global__ void init_state_kernel(__bf16* __restrict__ ah0, float* __restrict__ c) {
    unsigned id = blockIdx.x * 256u + threadIdx.x;   // < 131072
    ah0[id] = (__bf16)0.0f;
    c[id]   = 0.0f;
}

// ---- one LSTM time step: z = x_t@Wx + h@Wh + b ; gates ; c,h update ----
// B panels staged through LDS with double-buffered async copies (shared by the
// block's 8 waves); A fragments are per-wave coalesced global b128 loads.
__global__ __launch_bounds__(256) void lstm_step_kernel(
    const __bf16* __restrict__ ax_t,    // A frags of x_t: [kc][mt][lane][e]
    const __bf16* __restrict__ bx,      // B frags of Wx
    const __bf16* __restrict__ bh,      // B frags of Wh
    const __bf16* __restrict__ ah_in,   // A frags of h (prev)
    __bf16*       __restrict__ ah_out,  // A frags of h (new)
    float*        __restrict__ c,
    float*        __restrict__ hout,
    const float* __restrict__ bg, const float* __restrict__ bi,
    const float* __restrict__ bf_, const float* __restrict__ bo)
{
    __shared__ __align__(128) char lds_b[2 * GROUP_LDS_BYTES];   // 64 KB

    const unsigned tid  = threadIdx.x;
    const unsigned lane = tid & 31u;
    const unsigned mt   = tid >> 5;       // wave id = m-tile, 0..7
    const unsigned jt   = blockIdx.x;     // 0..63
    const unsigned n    = lane & 15u;
    const unsigned hi   = lane >> 4;
    const unsigned j    = jt * 16u + n;

    // stage-helper indices (one async b128 per thread per chunk)
    const unsigned sg   = tid >> 6;       // gate 0..3
    const unsigned sub  = tid & 63u;      // 16B slice within fragment
    const unsigned lds_base0 = (unsigned)(size_t)&lds_b[0];
    const unsigned lds_thr   = sg * 1024u + sub * 16u;

    // accumulators initialized with biases (bias depends only on column j)
    v8f acc0, acc1, acc2, acc3;
    float vb0 = bg[j], vb1 = bi[j], vb2 = bf_[j], vb3 = bo[j];
#pragma unroll
    for (int r = 0; r < 8; ++r) { acc0[r] = vb0; acc1[r] = vb1; acc2[r] = vb2; acc3[r] = vb3; }

    const v16bf* apx = (const v16bf*)ax_t  + (mt * 32u + lane);
    const v16bf* aph = (const v16bf*)ah_in + (mt * 32u + lane);

    // ---- prologue: stage group 0 (x panel) into buf 0 ----
    {
        const __bf16* src = bx;
#pragma unroll
        for (int cc = 0; cc < CHUNKS_PER_GROUP; ++cc) {
            const __bf16* g = src + (((unsigned)cc * 4u + sg) * (unsigned)JT + jt) * 512u + sub * 8u;
            async_g2l_b128(lds_base0 + (unsigned)cc * CHUNK_LDS_BYTES + lds_thr, g);
        }
    }

#pragma unroll 1
    for (int grp = 0; grp < NGROUPS; ++grp) {
        wait_asynccnt0();
        __syncthreads();   // buf[grp&1] fully written; buf[(grp+1)&1] free to overwrite

        // stage next group into the other buffer (overlaps with compute below)
        if (grp + 1 < NGROUPS) {
            unsigned kbase = (unsigned)grp * 8u;           // chunk base in h panel
            unsigned dst   = lds_base0 + ((unsigned)(grp + 1) & 1u) * GROUP_LDS_BYTES + lds_thr;
#pragma unroll
            for (int cc = 0; cc < CHUNKS_PER_GROUP; ++cc) {
                const __bf16* g = bh + (((kbase + (unsigned)cc) * 4u + sg) * (unsigned)JT + jt) * 512u + sub * 8u;
                async_g2l_b128(dst + (unsigned)cc * CHUNK_LDS_BYTES, g);
            }
        }

        // ---- compute this group: 8 chunks x 4 gate-WMMAs, B from LDS ----
        const v16bf* abase = (grp == 0) ? apx : (aph + (unsigned)(grp - 1) * (8u * (unsigned)MT * 32u));
        v16bf areg[CHUNKS_PER_GROUP];
#pragma unroll
        for (int cc = 0; cc < CHUNKS_PER_GROUP; ++cc)
            areg[cc] = abase[cc * (MT * 32)];

        const v16bf* lb = (const v16bf*)(lds_b + ((unsigned)grp & 1u) * GROUP_LDS_BYTES) + lane;
#pragma unroll
        for (int cc = 0; cc < CHUNKS_PER_GROUP; ++cc) {
            acc0 = __builtin_amdgcn_wmma_f32_16x16x32_bf16(false, areg[cc], false, lb[cc * 128 +  0], (short)0, acc0, false, false);
            acc1 = __builtin_amdgcn_wmma_f32_16x16x32_bf16(false, areg[cc], false, lb[cc * 128 + 32], (short)0, acc1, false, false);
            acc2 = __builtin_amdgcn_wmma_f32_16x16x32_bf16(false, areg[cc], false, lb[cc * 128 + 64], (short)0, acc2, false, false);
            acc3 = __builtin_amdgcn_wmma_f32_16x16x32_bf16(false, areg[cc], false, lb[cc * 128 + 96], (short)0, acc3, false, false);
        }
    }

    // ---- gates + cell/hidden update; scatter h into A-fragment layout ----
    // C/D layout: element r -> M = mt*16 + r + 8*(lane/16), N = j
    unsigned km = j & 31u;
    unsigned kc_o = j >> 5;
    unsigned bb, ee;
    if (km < 8u)       { bb = 0u; ee = km;       }
    else if (km < 16u) { bb = 1u; ee = km - 8u;  }
    else if (km < 24u) { bb = 0u; ee = km - 8u;  }
    else               { bb = 1u; ee = km - 16u; }

#pragma unroll
    for (int r = 0; r < 8; ++r) {
        unsigned mrow = (unsigned)r + 8u * hi;
        unsigned M    = mt * 16u + mrow;
        unsigned idx  = M * HD + j;
        float g  = tanhf(acc0[r]);
        float ii = sigf(acc1[r]);
        float ff = sigf(acc2[r]);
        float oo = sigf(acc3[r]);
        float cn = g * ii + c[idx] * ff;
        c[idx] = cn;
        float h = tanhf(cn) * oo;
        hout[idx] = h;
        unsigned lp = mrow + 16u * bb;   // destination lane slot in A layout
        ah_out[((kc_o * (unsigned)MT + mt) * 32u + lp) * 16u + ee] = (__bf16)h;
    }
}

// ---- tiny output projection: (128x1024) @ (1024x10) + b ----
__global__ void proj_kernel(const float* __restrict__ h, const float* __restrict__ Wp,
                            const float* __restrict__ bp, float* __restrict__ out) {
    unsigned id = blockIdx.x * 256u + threadIdx.x;
    if (id >= BS * NC) return;
    unsigned b = id / NC, cc = id % NC;
    float s = bp[cc];
    for (int j = 0; j < HD; ++j) s += h[b * HD + j] * Wp[j * NC + cc];
    out[id] = s;
}

extern "C" void kernel_launch(void* const* d_in, const int* in_sizes, int n_in,
                              void* d_out, int out_size, void* d_ws, size_t ws_size,
                              hipStream_t stream) {
    (void)in_sizes; (void)n_in; (void)out_size; (void)ws_size;
    const float* x   = (const float*)d_in[0];
    const float* Wgx = (const float*)d_in[1];
    const float* Wgh = (const float*)d_in[2];
    const float* bg  = (const float*)d_in[3];
    const float* Wix = (const float*)d_in[4];
    const float* Wih = (const float*)d_in[5];
    const float* bi  = (const float*)d_in[6];
    const float* Wfx = (const float*)d_in[7];
    const float* Wfh = (const float*)d_in[8];
    const float* bf  = (const float*)d_in[9];
    const float* Wox = (const float*)d_in[10];
    const float* Woh = (const float*)d_in[11];
    const float* bo  = (const float*)d_in[12];
    const float* Wph = (const float*)d_in[13];
    const float* bp  = (const float*)d_in[14];
    float* out = (float*)d_out;

    char* ws = (char*)d_ws;
    __bf16* ax = (__bf16*)(ws);
    __bf16* bh = (__bf16*)(ws + BH_OFF);
    __bf16* bx = (__bf16*)(ws + BX_OFF);
    __bf16* ah = (__bf16*)(ws + AH_OFF);   // two ping-pong buffers of AH_ELEMS
    float*  cc = (float*)(ws + C_OFF);
    float*  hf = (float*)(ws + H_OFF);

    pack_x_kernel <<<AX_ELEMS / 256, 256, 0, stream>>>(x, ax);
    pack_wb_kernel<<<BH_ELEMS / 256, 256, 0, stream>>>(Wgh, Wih, Wfh, Woh, bh);
    pack_wb_kernel<<<BX_ELEMS / 256, 256, 0, stream>>>(Wgx, Wix, Wfx, Wox, bx);
    init_state_kernel<<<CH_ELEMS / 256, 256, 0, stream>>>(ah, cc);

    for (int t = 0; t < SEQ; ++t) {
        const __bf16* ax_t   = ax + (size_t)t * (KC_X * MT * 32 * 16);
        const __bf16* ah_in  = ah + (size_t)(t & 1) * AH_ELEMS;
        __bf16*       ah_out = ah + (size_t)((t + 1) & 1) * AH_ELEMS;
        lstm_step_kernel<<<JT, 256, 0, stream>>>(ax_t, bx, bh, ah_in, ah_out,
                                                 cc, hf, bg, bi, bf, bo);
    }

    proj_kernel<<<(BS * NC + 255) / 256, 256, 0, stream>>>(hf, Wph, bp, out);
}
